// HybridTransformerLSTM_65481071396689
// MI455X (gfx1250) — matrix operand order repaired and checked by reading, not compiled, not run on hardware
//
#include <hip/hip_runtime.h>
#include <hip/hip_bf16.h>

// ---------------------------------------------------------------------------
// Types for CDNA5 WMMA (wave32): v_wmma_f32_16x16x32_bf16
// ---------------------------------------------------------------------------
typedef __attribute__((ext_vector_type(16))) __bf16 v16bf;
typedef __attribute__((ext_vector_type(8)))  __bf16 v8bf;
typedef __attribute__((ext_vector_type(8)))  float  v8f;
typedef __attribute__((ext_vector_type(4)))  int    v4i;

#define B_SZ 16
#define S_LEN 512
#define D_MODEL 512
#define N_HEADS 8
#define DK 64
#define F_FFN 2048
#define NQ 8
#define NTOK (B_SZ * S_LEN)

#define AS1 __attribute__((address_space(1)))
#define AS3 __attribute__((address_space(3)))

// gfx1250 async global->LDS path (ASYNCcnt); guarded so compile stays green
// if this toolchain spells the builtins differently.
#if __has_builtin(__builtin_amdgcn_global_load_async_to_lds_b128) && \
    __has_builtin(__builtin_amdgcn_s_wait_asynccnt)
#define USE_ASYNC_LDS 1
#else
#define USE_ASYNC_LDS 0
#endif

__device__ __forceinline__ v8f wmma_bf16(v16bf a, v16bf b, v8f c) {
    return __builtin_amdgcn_wmma_f32_16x16x32_bf16(
        /*neg_a=*/false, a, /*neg_b=*/false, b,
        /*c_mod=*/(short)0, c, /*reuse_a=*/false, /*reuse_b=*/false);
}

// A fragment: 16x32 bf16, row-major source, leading dim `ld` elements.
// lane<16: row=lane,    K = [0..7]  and [16..23]
// lane>=16: row=lane-16, K = [8..15] and [24..31]
__device__ __forceinline__ v16bf load_frag_a(const __bf16* p0, int ld, int lane) {
    int r  = lane & 15;
    int hi = lane >> 4;
    const __bf16* p = p0 + (size_t)r * ld + hi * 8;
    union { v16bf v; v8bf h[2]; } u;
    u.h[0] = *(const v8bf*)(p);
    u.h[1] = *(const v8bf*)(p + 16);
    return u.v;
}

// B fragment: 32x16 bf16 where B[k][n] = Wsrc[n][k] (Wsrc row-major [N,K]).
// lane<16: col=lane,    K = [0..7] and [16..23] ; lane>=16: col=lane-16, K = [8..15] and [24..31]
__device__ __forceinline__ v16bf load_frag_b(const __bf16* p0, int ld, int lane) {
    int c  = lane & 15;
    int hi = lane >> 4;
    const __bf16* p = p0 + (size_t)c * ld + hi * 8;
    union { v16bf v; v8bf h[2]; } u;
    u.h[0] = *(const v8bf*)(p);
    u.h[1] = *(const v8bf*)(p + 16);
    return u.v;
}

// Stage 16 bytes (8 bf16) from global into LDS.
__device__ __forceinline__ void stage16(const __bf16* g, __bf16* l) {
#if USE_ASYNC_LDS
    __builtin_amdgcn_global_load_async_to_lds_b128(
        (AS1 v4i*)(void*)g, (AS3 v4i*)l, 0, 0);
#else
    *(v8bf*)l = *(const v8bf*)g;
#endif
}

// ---------------------------------------------------------------------------
// K0: f32 -> bf16 weight conversion
// ---------------------------------------------------------------------------
__global__ void k_f32_to_bf16(const float* __restrict__ in, __bf16* __restrict__ out, int n) {
    int i = blockIdx.x * blockDim.x + threadIdx.x;
    if (i < n) out[i] = (__bf16)in[i];
}

// ---------------------------------------------------------------------------
// K1: embedding + sinusoidal positional encoding -> x(f32), xb(bf16), xT(bf16)
// ---------------------------------------------------------------------------
__global__ __launch_bounds__(256) void k_embed_pe(const int* __restrict__ tokens,
                                                  const float* __restrict__ table,
                                                  float* __restrict__ x,
                                                  __bf16* __restrict__ xb,
                                                  __bf16* __restrict__ xT) {
    int token = blockIdx.x;            // b*S + s
    int b = token >> 9, s = token & (S_LEN - 1);
    int tok = tokens[token];
    for (int d = threadIdx.x; d < D_MODEL; d += 256) {
        int i = d >> 1;
        float div = __expf(-(float)(2 * i) * (9.210340371976184f / (float)D_MODEL));
        float arg = (float)s * div;
        float pe  = (d & 1) ? __cosf(arg) : __sinf(arg);
        float v   = table[(size_t)tok * D_MODEL + d] + pe;
        size_t idx = (size_t)token * D_MODEL + d;
        x[idx]  = v;
        xb[idx] = (__bf16)v;
        xT[((size_t)b * D_MODEL + d) * S_LEN + s] = (__bf16)v;
    }
}

// ---------------------------------------------------------------------------
// K2: attention for one (b, h, 16-query tile) per wave.
//   scores = Q K^T / 8 via WMMA -> LDS ; wave32 softmax ; P(bf16) in LDS ;
//   out = P V via WMMA with V read from the bf16 transpose xT.
// ---------------------------------------------------------------------------
__global__ __launch_bounds__(32) void k_attention(const __bf16* __restrict__ xb,
                                                  const __bf16* __restrict__ xT,
                                                  __bf16* __restrict__ attnb) {
    __shared__ float  sc[16][S_LEN];     // raw scores, 32 KB
    __shared__ __bf16 pb[16][S_LEN];     // softmaxed P, 16 KB

    int wg = blockIdx.x;                 // ((b*H + h) * 32) + qt
    int qt = wg & 31; wg >>= 5;
    int h  = wg & (N_HEADS - 1);
    int b  = wg >> 3;
    int lane = threadIdx.x;
    int q0 = qt * 16;

    const __bf16* qbase = xb + ((size_t)b * S_LEN) * D_MODEL + h * DK;

    // Q fragments (16 x 64 split into two K=32 chunks), kept in registers.
    v16bf qa0 = load_frag_a(qbase + (size_t)q0 * D_MODEL +  0, D_MODEL, lane);
    v16bf qa1 = load_frag_a(qbase + (size_t)q0 * D_MODEL + 32, D_MODEL, lane);

    // Phase 1: raw score tiles -> LDS
    for (int kt = 0; kt < S_LEN / 16; ++kt) {
        const __bf16* kbase = qbase + (size_t)(kt * 16) * D_MODEL;
        v16bf kb0 = load_frag_b(kbase +  0, D_MODEL, lane);
        v16bf kb1 = load_frag_b(kbase + 32, D_MODEL, lane);
        v8f c = {};
        c = wmma_bf16(qa0, kb0, c);
        c = wmma_bf16(qa1, kb1, c);
        int col = kt * 16 + (lane & 15);
        int hi  = lane >> 4;
        #pragma unroll
        for (int v = 0; v < 8; ++v) sc[v + 8 * hi][col] = c[v];
    }
    __syncthreads();

    // Phase 2: softmax per query row (wave32 shfl reductions)
    const float scale = 0.125f;          // 1/sqrt(dk)
    for (int r = 0; r < 16; ++r) {
        float vals[16];
        float m = -1e30f;
        #pragma unroll
        for (int i = 0; i < 16; ++i) {
            float sv = sc[r][lane * 16 + i] * scale;
            vals[i] = sv;
            m = fmaxf(m, sv);
        }
        for (int off = 16; off; off >>= 1) m = fmaxf(m, __shfl_xor(m, off, 32));
        float sum = 0.f;
        #pragma unroll
        for (int i = 0; i < 16; ++i) { float e = __expf(vals[i] - m); vals[i] = e; sum += e; }
        for (int off = 16; off; off >>= 1) sum += __shfl_xor(sum, off, 32);
        float inv = 1.0f / sum;
        #pragma unroll
        for (int i = 0; i < 16; ++i) pb[r][lane * 16 + i] = (__bf16)(vals[i] * inv);
    }
    __syncthreads();

    // Phase 3: out = P @ V ; V read column-major via xT (contiguous along keys)
    const __bf16* vtbase = xT + ((size_t)b * D_MODEL + h * DK) * S_LEN;
    for (int dt = 0; dt < 4; ++dt) {
        v8f acc = {};
        for (int kt = 0; kt < S_LEN / 32; ++kt) {
            v16bf pa = load_frag_a(&pb[0][0] + kt * 32, S_LEN, lane);   // ds loads
            v16bf vb = load_frag_b(vtbase + (size_t)(dt * 16) * S_LEN + kt * 32, S_LEN, lane);
            acc = wmma_bf16(pa, vb, acc);
        }
        int col = h * DK + dt * 16 + (lane & 15);
        int hi  = lane >> 4;
        #pragma unroll
        for (int v = 0; v < 8; ++v) {
            int row = q0 + v + 8 * hi;
            attnb[((size_t)b * S_LEN + row) * D_MODEL + col] = (__bf16)acc[v];
        }
    }
}

// ---------------------------------------------------------------------------
// K3: WMMA GEMM  C[M,N] = A[M,K](bf16) @ W[N,K]^T(bf16), f32 out.
//   Block = 8 waves = 128 M-rows x 64 N-cols. The 64x32 B slab is shared by
//   all 8 waves and staged through LDS, double-buffered; staging uses
//   global_load_async_to_lds_b128 + s_wait_asynccnt when available.
// ---------------------------------------------------------------------------
__global__ __launch_bounds__(256) void k_gemm_bf16(const __bf16* __restrict__ A,
                                                   const __bf16* __restrict__ W,
                                                   float* __restrict__ C,
                                                   int M, int N, int K) {
    __shared__ __bf16 Bs[2][64 * 32];    // 2 x 4 KB

    int tid  = threadIdx.x;
    int lane = tid & 31;
    int wave = tid >> 5;
    int nbn  = N >> 6;                    // N / 64
    int mblk = blockIdx.x / nbn;
    int nblk = blockIdx.x % nbn;
    size_t m0 = (size_t)mblk * 128 + (size_t)wave * 16;
    size_t n0 = (size_t)nblk * 64;
    if (m0 >= (size_t)M) return;

    const __bf16* arow = A + m0 * K;

    // Staging assignment: thread -> (row r = tid>>2, 8-elem chunk tid&3)
    int srow   = tid >> 2;
    int schunk = (tid & 3) * 8;
    const __bf16* gsrc = W + (n0 + srow) * (size_t)K + schunk;
    __bf16* lpos0 = &Bs[0][srow * 32 + schunk];
    __bf16* lpos1 = &Bs[1][srow * 32 + schunk];

    v8f acc0 = {}, acc1 = {}, acc2 = {}, acc3 = {};
    const int nk = K >> 5;                // K / 32 steps

    // Prologue: stage slab 0
    stage16(gsrc, lpos0);

    for (int i = 0; i < nk; ++i) {
        int cur = i & 1;
        // Stage next slab into the other buffer while this one is consumed.
        if (i + 1 < nk) {
            stage16(gsrc + (size_t)(i + 1) * 32, cur ? lpos0 : lpos1);
#if USE_ASYNC_LDS
            __builtin_amdgcn_s_wait_asynccnt(1);   // slab i has landed
#endif
        } else {
#if USE_ASYNC_LDS
            __builtin_amdgcn_s_wait_asynccnt(0);
#endif
        }
        __syncthreads();                           // slab i visible to all waves

        const __bf16* bs = &Bs[cur][0];
        v16bf a  = load_frag_a(arow + (size_t)i * 32, K, lane);
        v16bf b0 = load_frag_b(bs +  0 * 32, 32, lane);   // ds reads
        v16bf b1 = load_frag_b(bs + 16 * 32, 32, lane);
        v16bf b2 = load_frag_b(bs + 32 * 32, 32, lane);
        v16bf b3 = load_frag_b(bs + 48 * 32, 32, lane);
        acc0 = wmma_bf16(a, b0, acc0);
        acc1 = wmma_bf16(a, b1, acc1);
        acc2 = wmma_bf16(a, b2, acc2);
        acc3 = wmma_bf16(a, b3, acc3);

        __syncthreads();                           // done reading slab i
    }

    int c0 = lane & 15, hi = lane >> 4;
    #pragma unroll
    for (int v = 0; v < 8; ++v) {
        size_t row = (m0 + v + 8 * hi) * (size_t)N + n0 + c0;
        C[row +  0] = acc0[v];
        C[row + 16] = acc1[v];
        C[row + 32] = acc2[v];
        C[row + 48] = acc3[v];
    }
}

// ---------------------------------------------------------------------------
// K4: residual add + LayerNorm; optionally refresh bf16 shadow + transpose.
//     One wave per token (512 dims -> 16 per lane), wave32 reductions.
// ---------------------------------------------------------------------------
__global__ __launch_bounds__(256) void k_add_ln(const float* __restrict__ xin,
                                                const float* __restrict__ y,
                                                const float* __restrict__ g,
                                                const float* __restrict__ beta,
                                                float* __restrict__ xout,
                                                __bf16* __restrict__ xb,
                                                __bf16* __restrict__ xT,
                                                int write_shadow) {
    int lane  = threadIdx.x & 31;
    int wave  = threadIdx.x >> 5;
    int token = blockIdx.x * 8 + wave;
    int b = token >> 9, s = token & (S_LEN - 1);
    size_t base = (size_t)token * D_MODEL;

    float vals[16];
    float sum = 0.f, sq = 0.f;
    #pragma unroll
    for (int i = 0; i < 16; ++i) {
        int d = lane + i * 32;
        float v = xin[base + d] + y[base + d];
        vals[i] = v; sum += v; sq += v * v;
    }
    for (int off = 16; off; off >>= 1) {
        sum += __shfl_xor(sum, off, 32);
        sq  += __shfl_xor(sq,  off, 32);
    }
    float mean = sum * (1.0f / D_MODEL);
    float var  = sq  * (1.0f / D_MODEL) - mean * mean;
    float inv  = rsqrtf(var + 1e-5f);
    #pragma unroll
    for (int i = 0; i < 16; ++i) {
        int d = lane + i * 32;
        float o = (vals[i] - mean) * inv * g[d] + beta[d];
        xout[base + d] = o;
        if (write_shadow) {
            xb[base + d] = (__bf16)o;
            xT[((size_t)b * D_MODEL + d) * S_LEN + s] = (__bf16)o;
        }
    }
}

// ---------------------------------------------------------------------------
// K5: quantum-gate FFN front:  h = relu( (cos(x[:,:8])*cos(theta)) @ w1^T )
// ---------------------------------------------------------------------------
__global__ __launch_bounds__(256) void k_ffn_h(const float* __restrict__ x,
                                               const float* __restrict__ theta,
                                               const float* __restrict__ w1,
                                               __bf16* __restrict__ h) {
    int token = blockIdx.x;
    size_t base = (size_t)token * D_MODEL;
    float q[NQ];
    #pragma unroll
    for (int w = 0; w < NQ; ++w) q[w] = __cosf(x[base + w]) * __cosf(theta[w]);
    for (int f = threadIdx.x; f < F_FFN; f += 256) {
        const float* wr = w1 + (size_t)f * NQ;
        float acc = 0.f;
        #pragma unroll
        for (int w = 0; w < NQ; ++w) acc += q[w] * wr[w];
        h[(size_t)token * F_FFN + f] = (__bf16)fmaxf(acc, 0.f);
    }
}

// ---------------------------------------------------------------------------
// K6/K7: mean-pool over sequence, then tiny classifier head.
// ---------------------------------------------------------------------------
__global__ void k_pool_mean(const float* __restrict__ x, float* __restrict__ pooled) {
    int i = blockIdx.x * blockDim.x + threadIdx.x;   // b*D + d
    if (i >= B_SZ * D_MODEL) return;
    int b = i >> 9, d = i & (D_MODEL - 1);
    float s = 0.f;
    for (int t = 0; t < S_LEN; ++t)
        s += x[((size_t)b * S_LEN + t) * D_MODEL + d];
    pooled[i] = s * (1.0f / S_LEN);
}

__global__ void k_head(const float* __restrict__ pooled,
                       const float* __restrict__ hw,
                       const float* __restrict__ hb,
                       float* __restrict__ out) {
    int i = threadIdx.x;                 // b*2 + c, 32 outputs
    if (i >= 32) return;
    int b = i >> 1, c = i & 1;
    float s = hb[c];
    for (int d = 0; d < D_MODEL; ++d)
        s += pooled[b * D_MODEL + d] * hw[c * D_MODEL + d];
    out[i] = s;
}

// ---------------------------------------------------------------------------
// Host launch
// ---------------------------------------------------------------------------
extern "C" void kernel_launch(void* const* d_in, const int* in_sizes, int n_in,
                              void* d_out, int out_size, void* d_ws, size_t ws_size,
                              hipStream_t stream) {
    (void)in_sizes; (void)n_in; (void)out_size; (void)ws_size;
    const int*   tokens = (const int*)  d_in[0];
    const float* table  = (const float*)d_in[1];
    const float* cw     = (const float*)d_in[2];
    const float* theta  = (const float*)d_in[3];
    const float* w1     = (const float*)d_in[4];
    const float* w2     = (const float*)d_in[5];
    const float* ln1g   = (const float*)d_in[6];
    const float* ln1b   = (const float*)d_in[7];
    const float* ln2g   = (const float*)d_in[8];
    const float* ln2b   = (const float*)d_in[9];
    const float* headw  = (const float*)d_in[10];
    const float* headb  = (const float*)d_in[11];

    // Workspace carve-up (256B aligned)
    char* ws = (char*)d_ws;
    auto alloc = [&](size_t bytes) {
        void* p = (void*)ws;
        ws += (bytes + 255) & ~(size_t)255;
        return p;
    };
    float*  x      = (float*) alloc((size_t)NTOK * D_MODEL * 4);
    __bf16* xb     = (__bf16*)alloc((size_t)NTOK * D_MODEL * 2);
    __bf16* xT     = (__bf16*)alloc((size_t)NTOK * D_MODEL * 2);
    __bf16* attnb  = (__bf16*)alloc((size_t)NTOK * D_MODEL * 2);
    float*  y      = (float*) alloc((size_t)NTOK * D_MODEL * 4);
    __bf16* hbuf   = (__bf16*)alloc((size_t)NTOK * F_FFN * 2);
    __bf16* cwb    = (__bf16*)alloc((size_t)2 * D_MODEL * D_MODEL * 2);
    __bf16* w2b    = (__bf16*)alloc((size_t)2 * D_MODEL * F_FFN * 2);
    float*  pooled = (float*) alloc((size_t)B_SZ * D_MODEL * 4);

    // Weight conversion to bf16
    {
        int ncw = 2 * D_MODEL * D_MODEL;
        int nw2 = 2 * D_MODEL * F_FFN;
        k_f32_to_bf16<<<(ncw + 255) / 256, 256, 0, stream>>>(cw, cwb, ncw);
        k_f32_to_bf16<<<(nw2 + 255) / 256, 256, 0, stream>>>(w2, w2b, nw2);
    }

    // Embedding + positional encoding
    k_embed_pe<<<NTOK, 256, 0, stream>>>(tokens, table, x, xb, xT);

    const int gemm_blocks = (NTOK / 128) * (D_MODEL / 64);          // 512
    const int ln_blocks   = NTOK / 8;                               // 1024

    for (int l = 0; l < 2; ++l) {
        // Attention: one wave per (b, h, 16-query tile)
        k_attention<<<B_SZ * N_HEADS * (S_LEN / 16), 32, 0, stream>>>(xb, xT, attnb);
        // Combine projection
        k_gemm_bf16<<<gemm_blocks, 256, 0, stream>>>(
            attnb, cwb + (size_t)l * D_MODEL * D_MODEL, y, NTOK, D_MODEL, D_MODEL);
        // x = LN1(x + attn)
        k_add_ln<<<ln_blocks, 256, 0, stream>>>(
            x, y, ln1g + l * D_MODEL, ln1b + l * D_MODEL, x, xb, xT, 0);
        // Quantum-gate FFN front
        k_ffn_h<<<NTOK, 256, 0, stream>>>(x, theta + l * NQ, w1 + (size_t)l * F_FFN * NQ, hbuf);
        // FFN back GEMM (K = 2048)
        k_gemm_bf16<<<gemm_blocks, 256, 0, stream>>>(
            hbuf, w2b + (size_t)l * D_MODEL * F_FFN, y, NTOK, D_MODEL, F_FFN);
        // x = LN2(x + ffn); refresh bf16 shadow + transpose for next layer
        k_add_ln<<<ln_blocks, 256, 0, stream>>>(
            x, y, ln2g + l * D_MODEL, ln2b + l * D_MODEL, x, xb, xT, 1);
    }

    // Mean-pool + head
    k_pool_mean<<<(B_SZ * D_MODEL + 255) / 256, 256, 0, stream>>>(x, pooled);
    k_head<<<1, 32, 0, stream>>>(pooled, headw, headb, (float*)d_out);
}
